// torchLSTM_cell_mc_44598940402206
// MI455X (gfx1250) — matrix operand order repaired and checked by reading, not compiled, not run on hardware
//
#include <hip/hip_runtime.h>
#include <hip/hip_bf16.h>

// ---------------------------------------------------------------------------
// LSTM for MI455X (gfx1250):
//   Kernel prep : swizzle W_in / W_ih / W_hh (f32) into f16 WMMA B-fragments
//   Kernel xg   : xg = relu(x@W_in^T + b_in) @ W_ih^T + b_ih + b_hh   (WMMA)
//                 stored f16 in C-fragment order (1.53 GB workspace)
//   Kernel rec  : 365 sequential steps; 128 independent workgroups of 16 rows;
//                 h in LDS (f16, padded rows), c in accumulator VGPRs;
//                 gates via V_WMMA_F32_16X16X32_F16; out = h@W_out^T + b_out
// ---------------------------------------------------------------------------

typedef __attribute__((ext_vector_type(16))) _Float16 v16h;
typedef __attribute__((ext_vector_type(8)))  _Float16 v8h;
typedef __attribute__((ext_vector_type(2)))  _Float16 hpair;
typedef __attribute__((ext_vector_type(8)))  float    v8f;

#define NT     365
#define NGRID  2048
#define NX     32
#define NH     256
#define NG     1024            // 4*NH
#define MROWS  (NT * NGRID)    // 747520
#define MTILES (MROWS / 16)    // 46720
#define NTILES (NG / 16)       // 64
#define KSTEPS (NH / 32)       // 8
#define HS     264             // padded f16 row stride for h/x0 in LDS

__device__ __forceinline__ float sigm(float x) {
    return 1.0f / (1.0f + __expf(-x));
}
__device__ __forceinline__ float tanh_(float x) {
    float e = __expf(2.0f * x);
    return (e - 1.0f) / (e + 1.0f);
}

// A-fragment (16x32 f16) from an LDS row-major buffer, contiguous k order (matches prep_bfrag):
// lanes 0-15: rows 0-15, K 0..15; lanes 16-31: rows 0-15, K 16..31
__device__ __forceinline__ v16h load_a_frag(const _Float16* buf, int ld, int kbase, int lane) {
    int row = lane & 15;
    int kg  = (lane >> 4) << 4;   // 0 or 16
    const _Float16* rp = buf + row * ld + kbase + kg;
    v16h a;
    hpair* ap = (hpair*)&a;
#pragma unroll
    for (int p = 0; p < 4; ++p) ap[p]     = *(const hpair*)(rp + 2 * p);
#pragma unroll
    for (int p = 0; p < 4; ++p) ap[p + 4] = *(const hpair*)(rp + 8 + 2 * p);
    return a;
}

// --------------------------------------------------------------------------
// Swizzle f32 weight matrix W [ntiles*16, ld] into f16 B-fragments (K x N):
// frag index ((ntile*ksteps + ks)*32 + lane)*16 ; lanes 0-15 hold K 0..15 of
// the k-step, lanes 16-31 hold K 16..31; column n = ntile*16 + (lane&15).
// --------------------------------------------------------------------------
__global__ void prep_bfrag(const float* __restrict__ W, _Float16* __restrict__ dst,
                           int ntiles, int ksteps, int ld) {
    int tid   = blockIdx.x * blockDim.x + threadIdx.x;
    int total = ntiles * ksteps * 32;
    if (tid >= total) return;
    int lane  = tid & 31;
    int ks    = (tid >> 5) % ksteps;
    int ntile = tid / (32 * ksteps);
    int n     = ntile * 16 + (lane & 15);
    int kg    = (lane >> 4) << 4;   // 0 or 16
    _Float16* o = dst + (size_t)tid * 16;
#pragma unroll
    for (int p = 0; p < 8; ++p) {
        int K = ks * 32 + kg + 2 * p;
        o[2 * p]     = (_Float16)W[(size_t)n * ld + K];
        o[2 * p + 1] = (_Float16)W[(size_t)n * ld + K + 1];
    }
}

// --------------------------------------------------------------------------
// Kernel A: xg = relu(x@W_in^T + b_in) @ W_ih^T + (b_ih + b_hh), f16,
// stored in C-fragment order: ((rowtile*64 + ntile)*32 + lane)*8 halves.
// One 16-row tile per 256-thread (8-wave) workgroup.
// --------------------------------------------------------------------------
__global__ __launch_bounds__(256) void lstm_xg_kernel(
    const float* __restrict__ x, const float* __restrict__ b_in,
    const _Float16* __restrict__ win_frag, const _Float16* __restrict__ wih_frag,
    const float* __restrict__ b_ih, const float* __restrict__ b_hh,
    _Float16* __restrict__ xg) {
    __shared__ _Float16 xls[16 * 40];    // 16x32 input tile, padded
    __shared__ _Float16 x0ls[16 * HS];   // 16x256 relu tile, padded
    const int tid  = threadIdx.x;
    const int lane = tid & 31;
    const int w    = tid >> 5;
    const int nsel = lane & 15;
    const int mhi  = (lane >> 4) << 3;
    const size_t r0 = (size_t)blockIdx.x * 16;

    // stage 0: load + convert 16x32 x tile to f16 LDS
    for (int i = tid; i < 512; i += 256) {
        int m = i >> 5, k = i & 31;
        xls[m * 40 + k] = (_Float16)x[(r0 + m) * NX + k];
    }
    __syncthreads();

    // stage 1: x0 = relu(x @ W_in^T + b_in), K = 32 (single WMMA per n-tile)
    v16h ax = load_a_frag(xls, 40, 0, lane);
#pragma unroll
    for (int s = 0; s < 2; ++s) {
        int ntile = w * 2 + s;                       // 16 n-tiles over 8 waves
        v16h bf = *(const v16h*)(win_frag + ((size_t)ntile * 32 + lane) * 16);
        v8f acc = {};
        acc = __builtin_amdgcn_wmma_f32_16x16x32_f16(false, ax, false, bf,
                                                     (short)0, acc, false, false);
        int n    = ntile * 16 + nsel;
        float bb = b_in[n];
#pragma unroll
        for (int e = 0; e < 8; ++e) {
            float v = acc[e] + bb;
            v = v > 0.0f ? v : 0.0f;
            x0ls[(e + mhi) * HS + n] = (_Float16)v;
        }
    }
    __syncthreads();

    // stage 2: xg = x0 @ W_ih^T + b ; 64 n-tiles over 8 waves, K = 256
    v16h afr[KSTEPS];
#pragma unroll
    for (int ks = 0; ks < KSTEPS; ++ks) afr[ks] = load_a_frag(x0ls, HS, ks * 32, lane);
    const size_t tilebase = (size_t)blockIdx.x * NTILES;
#pragma unroll
    for (int q = 0; q < 8; ++q) {
        int ntile = w * 8 + q;
        const _Float16* bp = wih_frag + (size_t)ntile * KSTEPS * 32 * 16;
        v8f acc = {};
#pragma unroll
        for (int ks = 0; ks < KSTEPS; ++ks) {
            v16h bf = *(const v16h*)(bp + ((size_t)ks * 32 + lane) * 16);
            acc = __builtin_amdgcn_wmma_f32_16x16x32_f16(false, afr[ks], false, bf,
                                                         (short)0, acc, false, false);
        }
        int n    = ntile * 16 + nsel;
        float bb = b_ih[n] + b_hh[n];
        v8h o;
#pragma unroll
        for (int e = 0; e < 8; ++e) o[e] = (_Float16)(acc[e] + bb);
        *(v8h*)(xg + (tilebase + ntile) * 256 + (size_t)lane * 8) = o;
    }
}

// --------------------------------------------------------------------------
// Kernel B: the recurrence. 128 independent workgroups x 16 grid rows.
// Per step per wave: 64 x V_WMMA_F32_16X16X32_F16 (wave w owns hidden slice
// j in [32w, 32w+32) => gate n-tiles {g*16 + 2w, g*16 + 2w + 1}).
// c stays in accumulator-layout VGPRs; h round-trips through padded LDS.
// --------------------------------------------------------------------------
__global__ __launch_bounds__(256) void lstm_recurrent_kernel(
    const _Float16* __restrict__ xg, const _Float16* __restrict__ whh_frag,
    const float* __restrict__ W_out, const float* __restrict__ b_out,
    float* __restrict__ out) {
    __shared__ _Float16 h16[16 * HS];
    __shared__ float wout[NH];
    const int tid  = threadIdx.x;
    const int lane = tid & 31;
    const int w    = tid >> 5;
    const int nsel = lane & 15;
    const int mhi  = (lane >> 4) << 3;
    const int blk  = blockIdx.x;
    const int g0   = blk * 16;

    for (int i = tid; i < 16 * HS; i += 256) h16[i] = (_Float16)0.0f;
    wout[tid] = W_out[tid];
    const float bo = b_out[0];
    v8f c0 = {}, c1 = {};
    __syncthreads();

    for (int t = 0; t < NT; ++t) {
        // A fragments of h (shared across all 8 gate tiles of this wave)
        v16h afr[KSTEPS];
#pragma unroll
        for (int ks = 0; ks < KSTEPS; ++ks) afr[ks] = load_a_frag(h16, HS, ks * 32, lane);

        const size_t tilebase = ((size_t)t * (NGRID / 16) + blk) * NTILES;
        v8f acc[8];
#pragma unroll
        for (int q = 0; q < 8; ++q) {
            int gate  = q >> 1, sub = q & 1;
            int ntile = gate * 16 + w * 2 + sub;
            // init accumulator from precomputed xg (C-fragment order, 16B/lane)
            const _Float16* xp = xg + (tilebase + ntile) * 256 + (size_t)lane * 8;
            __builtin_prefetch(xp + (size_t)(NGRID / 16) * NTILES * 256, 0, 0); // next t
            v8h xv = *(const v8h*)xp;
            v8f a;
#pragma unroll
            for (int e = 0; e < 8; ++e) a[e] = (float)xv[e];
            const _Float16* bp = whh_frag + (size_t)ntile * KSTEPS * 32 * 16;
#pragma unroll
            for (int ks = 0; ks < KSTEPS; ++ks) {
                v16h bf = *(const v16h*)(bp + ((size_t)ks * 32 + lane) * 16);
                a = __builtin_amdgcn_wmma_f32_16x16x32_f16(false, afr[ks], false, bf,
                                                           (short)0, a, false, false);
            }
            acc[q] = a;
        }
        __syncthreads();   // everyone done reading h16

        // elementwise LSTM cell update directly in C-fragment registers
#pragma unroll
        for (int sub = 0; sub < 2; ++sub) {
            v8f& cc = sub ? c1 : c0;
#pragma unroll
            for (int e = 0; e < 8; ++e) {
                float iv = sigm(acc[0 + sub][e]);
                float fv = sigm(acc[2 + sub][e]);
                float gv = tanh_(acc[4 + sub][e]);
                float ov = sigm(acc[6 + sub][e]);
                float cv = fv * cc[e] + iv * gv;
                cc[e] = cv;
                float hv = ov * tanh_(cv);
                h16[(e + mhi) * HS + (w * 32 + sub * 16 + nsel)] = (_Float16)hv;
            }
        }
        __syncthreads();   // h16 fully updated

        // out[t, g] = h . W_out + b_out   (NY == 1; 16 rows -> 16 lanes)
        if (tid < 16) {
            float s = 0.0f;
#pragma unroll 8
            for (int j = 0; j < NH; ++j) s += (float)h16[tid * HS + j] * wout[j];
            out[(size_t)t * NGRID + g0 + tid] = s + bo;
        }
    }
}

extern "C" void kernel_launch(void* const* d_in, const int* in_sizes, int n_in,
                              void* d_out, int out_size, void* d_ws, size_t ws_size,
                              hipStream_t stream) {
    const float* x     = (const float*)d_in[0];
    const float* W_in  = (const float*)d_in[1];
    const float* b_in  = (const float*)d_in[2];
    const float* W_ih  = (const float*)d_in[3];
    const float* b_ih  = (const float*)d_in[4];
    const float* W_hh  = (const float*)d_in[5];
    const float* b_hh  = (const float*)d_in[6];
    const float* W_out = (const float*)d_in[7];
    const float* b_out = (const float*)d_in[8];
    float* out = (float*)d_out;

    // workspace layout (halves): [xg | whh_frag | wih_frag | win_frag]
    const size_t XG_HALVES = (size_t)MTILES * NTILES * 256;         // 765,460,480
    const size_t WFRAG     = (size_t)NTILES * KSTEPS * 32 * 16;     // 262,144
    _Float16* xg_ws  = (_Float16*)d_ws;
    _Float16* whh_fr = xg_ws + XG_HALVES;
    _Float16* wih_fr = whh_fr + WFRAG;
    _Float16* win_fr = wih_fr + WFRAG;

    prep_bfrag<<<2, 256, 0, stream>>>(W_in, win_fr, 16, 1, NX);
    prep_bfrag<<<(int)(WFRAG / 16 + 255) / 256, 256, 0, stream>>>(W_ih, wih_fr, NTILES, KSTEPS, NH);
    prep_bfrag<<<(int)(WFRAG / 16 + 255) / 256, 256, 0, stream>>>(W_hh, whh_fr, NTILES, KSTEPS, NH);

    lstm_xg_kernel<<<MTILES, 256, 0, stream>>>(x, b_in, win_fr, wih_fr, b_ih, b_hh, xg_ws);
    lstm_recurrent_kernel<<<NGRID / 16, 256, 0, stream>>>(xg_ws, whh_fr, W_out, b_out, out);
}
